// GNN_Aval_76605036692112
// MI455X (gfx1250) — compile-verified, matrix operand order checked by reading, not run on hardware
//
#include <hip/hip_runtime.h>
#include <cmath>

#define NT 20000
#define NV 512
#define EDEP 150000
#define ECAN 150000
#define HID 256
#define HEADS 4
#define CPH 64
#define LAYERS 4
#define NEG 0.2f

typedef __attribute__((ext_vector_type(16))) __bf16 v16bf;
typedef __attribute__((ext_vector_type(8)))  __bf16 v8bf;
typedef __attribute__((ext_vector_type(8)))  float  v8f;

static __device__ __forceinline__ __bf16 f2bf(float f) {
  union { float f; unsigned u; } v; v.f = f;
  unsigned r = v.u + 0x7FFFu + ((v.u >> 16) & 1u);
  unsigned short h = (unsigned short)(r >> 16);
  __bf16 b; __builtin_memcpy(&b, &h, 2);
  return b;
}

static __device__ __forceinline__ void atomicMaxF(float* addr, float v) {
  if (v >= 0.f) atomicMax((int*)addr, __float_as_int(v));
  else          atomicMin((unsigned int*)addr, __float_as_uint(v));
}

// ---------------- input transforms ----------------
__global__ void k_input_task(const float* __restrict__ x, const float* __restrict__ W,
                             const float* __restrict__ b, float* __restrict__ h) {
  int i = blockIdx.x * blockDim.x + threadIdx.x;
  if (i >= NT * HID) return;
  int n = i / HID, f = i % HID;
  float acc = b[f];
#pragma unroll
  for (int k = 0; k < 8; ++k) acc += x[n * 8 + k] * W[k * HID + f];
  h[i] = fmaxf(acc, 0.f);
}

__global__ void k_input_vm(const float* __restrict__ x, const float* __restrict__ W,
                           const float* __restrict__ b, float* __restrict__ h) {
  int i = blockIdx.x * blockDim.x + threadIdx.x;
  if (i >= NV * HID) return;
  int n = i / HID, f = i % HID;
  float acc = b[f];
#pragma unroll
  for (int k = 0; k < 4; ++k) acc += x[n * 4 + k] * W[k * HID + f];
  h[i] = fmaxf(acc, 0.f);
}

// ---------------- conversions ----------------
__global__ void k_f32_to_bf16(const float* __restrict__ in, __bf16* __restrict__ out, int n) {
  int i = blockIdx.x * blockDim.x + threadIdx.x;
  if (i < n) out[i] = f2bf(in[i]);
}

// W [K=256][N=256] row-major fp32 -> Bt [N][K] bf16 (column-of-W contiguous)
__global__ void k_pack_w(const float* __restrict__ W, __bf16* __restrict__ Bt) {
  int i = blockIdx.x * blockDim.x + threadIdx.x;
  if (i >= HID * HID) return;
  int n = i / HID, k = i % HID;
  Bt[n * HID + k] = f2bf(W[k * HID + n]);
}

// ---------------- WMMA GEMM: C[M x 256] = A[M x 256] * W ----------------
// A bf16 row-major, Bt bf16 N-major (Bt[n*256+k] = W[k][n]).
// 128 threads = 4 waves; wave w computes a 16x64 strip (4 accumulators),
// reusing one A fragment across 4 WMMAs per K-step. grid.x = M/16.
__global__ __launch_bounds__(128) void k_gemm_wmma(
    const __bf16* __restrict__ A, const __bf16* __restrict__ Bt,
    float* __restrict__ C) {
  int wave = threadIdx.x >> 5;            // 0..3
  int lane = threadIdx.x & 31;
  int mBase = blockIdx.x * 16;
  int nBase = wave * 64;                  // 4 n-tiles of 16
  int hf  = lane >> 4;                    // half-wave select
  int l16 = lane & 15;
  const __bf16* arow = A + (size_t)(mBase + l16) * HID;
  const __bf16* brow = Bt + (size_t)(nBase + l16) * HID;  // tile t at +t*16*HID
  v8f acc0 = {}, acc1 = {}, acc2 = {}, acc3 = {};
#pragma unroll
  for (int kt = 0; kt < HID / 32; ++kt) {
    int k0 = kt * 32 + hf * 8;
    v8bf a0 = *(const v8bf*)(arow + k0);
    v8bf a1 = *(const v8bf*)(arow + k0 + 16);
    v16bf a;
#pragma unroll
    for (int i = 0; i < 8; ++i) { a[i] = a0[i]; a[i + 8] = a1[i]; }
#pragma unroll
    for (int t = 0; t < 4; ++t) {
      const __bf16* bp = brow + (size_t)t * 16 * HID;
      v8bf b0 = *(const v8bf*)(bp + k0);
      v8bf b1 = *(const v8bf*)(bp + k0 + 16);
      v16bf b;
#pragma unroll
      for (int i = 0; i < 8; ++i) { b[i] = b0[i]; b[i + 8] = b1[i]; }
      if (t == 0) acc0 = __builtin_amdgcn_wmma_f32_16x16x32_bf16(false, a, false, b, (short)0, acc0, false, false);
      if (t == 1) acc1 = __builtin_amdgcn_wmma_f32_16x16x32_bf16(false, a, false, b, (short)0, acc1, false, false);
      if (t == 2) acc2 = __builtin_amdgcn_wmma_f32_16x16x32_bf16(false, a, false, b, (short)0, acc2, false, false);
      if (t == 3) acc3 = __builtin_amdgcn_wmma_f32_16x16x32_bf16(false, a, false, b, (short)0, acc3, false, false);
    }
  }
  int m0 = mBase + hf * 8;
  float* crow = C + (size_t)m0 * HID + nBase + l16;
#pragma unroll
  for (int j = 0; j < 8; ++j) {
    crow[(size_t)j * HID +  0] = acc0[j];
    crow[(size_t)j * HID + 16] = acc1[j];
    crow[(size_t)j * HID + 32] = acc2[j];
    crow[(size_t)j * HID + 48] = acc3[j];
  }
}

// ---------------- attention logits: al[n,h] = dot(HS[n, h*64 : h*64+64], av[h]) ----------------
__global__ void k_attn(const float* __restrict__ HS, const float* __restrict__ av,
                       float* __restrict__ al, int N) {
  int i = blockIdx.x * blockDim.x + threadIdx.x;
  if (i >= N * HEADS) return;
  int n = i >> 2, h = i & 3;
  const float* row = HS + (size_t)n * HID + h * CPH;
  const float* a = av + h * CPH;
  float s = 0.f;
#pragma unroll 8
  for (int c = 0; c < CPH; ++c) s += row[c] * a[c];
  al[i] = s;
}

// ---------------- misc fills ----------------
__global__ void k_fill(float* __restrict__ p, float v, int n) {
  int i = blockIdx.x * blockDim.x + threadIdx.x;
  if (i < n) p[i] = v;
}
__global__ void k_init3(float* __restrict__ u, const float* __restrict__ ba,
                        const float* __restrict__ bb, const float* __restrict__ bc, int N) {
  int i = blockIdx.x * blockDim.x + threadIdx.x;
  if (i >= N * HID) return;
  int f = i % HID;
  u[i] = ba[f] + bb[f] + bc[f];
}
__global__ void k_init1(float* __restrict__ u, const float* __restrict__ b, int N) {
  int i = blockIdx.x * blockDim.x + threadIdx.x;
  if (i >= N * HID) return;
  u[i] = b[i % HID];
}

// ---------------- edge softmax passes (self loops = virtual edges >= E) ----------------
__global__ void k_edge_max(const int* __restrict__ src, const int* __restrict__ dst,
                           int E, int nLoop,
                           const float* __restrict__ als, const float* __restrict__ ald,
                           float* __restrict__ m) {
  int t = blockIdx.x * blockDim.x + threadIdx.x;
  if (t >= (E + nLoop) * HEADS) return;
  int e = t >> 2, h = t & 3;
  int s, d;
  if (e < E) { s = src[e]; d = dst[e]; } else { s = e - E; d = s; }
  float x = als[s * HEADS + h] + ald[d * HEADS + h];
  x = x > 0.f ? x : NEG * x;
  atomicMaxF(&m[d * HEADS + h], x);
}

__global__ void k_edge_expsum(const int* __restrict__ src, const int* __restrict__ dst,
                              int E, int nLoop,
                              const float* __restrict__ als, const float* __restrict__ ald,
                              const float* __restrict__ m, float* __restrict__ den) {
  int t = blockIdx.x * blockDim.x + threadIdx.x;
  if (t >= (E + nLoop) * HEADS) return;
  int e = t >> 2, h = t & 3;
  int s, d;
  if (e < E) { s = src[e]; d = dst[e]; } else { s = e - E; d = s; }
  float x = als[s * HEADS + h] + ald[d * HEADS + h];
  x = x > 0.f ? x : NEG * x;
  atomicAdd(&den[d * HEADS + h], __expf(x - m[d * HEADS + h]));
}

// one thread per (edge, head, 16-channel chunk): 4x float4 coalesced reads, 16 atomics
__global__ void k_edge_scatter(const int* __restrict__ src, const int* __restrict__ dst,
                               int E, int nLoop,
                               const float* __restrict__ als, const float* __restrict__ ald,
                               const float* __restrict__ m, const float* __restrict__ den,
                               const float* __restrict__ HS, float* __restrict__ U) {
  int t = blockIdx.x * blockDim.x + threadIdx.x;
  if (t >= (E + nLoop) * HEADS * 4) return;
  int c16 = t & 3;              // which 16-channel chunk
  int h   = (t >> 2) & 3;
  int e   = t >> 4;
  int s, d;
  if (e < E) { s = src[e]; d = dst[e]; } else { s = e - E; d = s; }
  int dh = d * HEADS + h;
  float x = als[s * HEADS + h] + ald[dh];
  x = x > 0.f ? x : NEG * x;
  float alpha = __expf(x - m[dh]) / den[dh];
  const float4* hr = (const float4*)(HS + (size_t)s * HID + h * CPH + c16 * 16);
  float* ur = U + (size_t)d * HID + h * CPH + c16 * 16;
#pragma unroll
  for (int q = 0; q < 4; ++q) {
    float4 v = hr[q];
    atomicAdd(ur + q * 4 + 0, alpha * v.x);
    atomicAdd(ur + q * 4 + 1, alpha * v.y);
    atomicAdd(ur + q * 4 + 2, alpha * v.z);
    atomicAdd(ur + q * 4 + 3, alpha * v.w);
  }
}

// ---------------- residual + relu combine ----------------
__global__ void k_combine(float* __restrict__ h, const float* __restrict__ u, int n) {
  int i = blockIdx.x * blockDim.x + threadIdx.x;
  if (i < n) h[i] += fmaxf(u[i], 0.f);
}

// ---------------- mean pool + head MLP ----------------
__global__ void k_pool(const float* __restrict__ ht, float* __restrict__ pooled) {
  int f = threadIdx.x;                // 256 threads, one feature each
  int r0 = blockIdx.x * 128;
  int r1 = r0 + 128; if (r1 > NT) r1 = NT;
  float s = 0.f;
  for (int r = r0; r < r1; ++r) s += ht[(size_t)r * HID + f];
  atomicAdd(&pooled[f], s);
}

__global__ void k_mlp(const float* __restrict__ pooled,
                      const float* __restrict__ W1, const float* __restrict__ b1_,
                      const float* __restrict__ W2, const float* __restrict__ b2_,
                      float* __restrict__ out) {
  __shared__ float p[HID];
  __shared__ float hbuf[HID / 2];
  int t = threadIdx.x;                // 128 threads
  p[t]       = pooled[t]       * (1.f / NT);
  p[t + 128] = pooled[t + 128] * (1.f / NT);
  __syncthreads();
  float acc = b1_[t];
  for (int k = 0; k < HID; ++k) acc += p[k] * W1[k * (HID / 2) + t];
  hbuf[t] = fmaxf(acc, 0.f);
  __syncthreads();
  if (t < 16) {
    float o = b2_[t];
    for (int j = 0; j < HID / 2; ++j) o += hbuf[j] * W2[j * 16 + t];
    out[t] = o;
  }
}

static inline int cdiv(int a, int b) { return (a + b - 1) / b; }

extern "C" void kernel_launch(void* const* d_in, const int* in_sizes, int n_in,
                              void* d_out, int out_size, void* d_ws, size_t ws_size,
                              hipStream_t stream) {
  (void)in_sizes; (void)n_in; (void)out_size; (void)ws_size;

  const float* x_task  = (const float*)d_in[0];
  const float* x_vm    = (const float*)d_in[1];
  const int*   ed      = (const int*)d_in[2];        // [2][EDEP]
  const int*   can_src = (const int*)d_in[3];
  const int*   can_dst = (const int*)d_in[4];
  const float* Wt_in   = (const float*)d_in[5];
  const float* bt      = (const float*)d_in[6];
  const float* Wv_in   = (const float*)d_in[7];
  const float* bv      = (const float*)d_in[8];
  const float* W1      = (const float*)d_in[9];
  const float* b1      = (const float*)d_in[10];
  const float* W2      = (const float*)d_in[11];
  const float* b2      = (const float*)d_in[12];
  const float* dep_W   = (const float*)d_in[13];
  const float* dep_as  = (const float*)d_in[14];
  const float* dep_ad  = (const float*)d_in[15];
  const float* dep_b   = (const float*)d_in[16];
  const float* rev_W   = (const float*)d_in[17];
  const float* rev_as  = (const float*)d_in[18];
  const float* rev_ad  = (const float*)d_in[19];
  const float* rev_b   = (const float*)d_in[20];
  const float* can_Ws  = (const float*)d_in[21];
  const float* can_Wd  = (const float*)d_in[22];
  const float* can_as  = (const float*)d_in[23];
  const float* can_ad  = (const float*)d_in[24];
  const float* can_b   = (const float*)d_in[25];
  const float* rcan_Ws = (const float*)d_in[26];
  const float* rcan_Wd = (const float*)d_in[27];
  const float* rcan_as = (const float*)d_in[28];
  const float* rcan_ad = (const float*)d_in[29];
  const float* rcan_b  = (const float*)d_in[30];

  const int* d_src = ed;
  const int* d_dst = ed + EDEP;

  // workspace carve-out
  char* w = (char*)d_ws;
  size_t off = 0;
  auto alloc = [&](size_t bytes) -> void* {
    off = (off + 255) & ~(size_t)255;
    void* p = w + off;
    off += bytes;
    return p;
  };
  float*  ht     = (float*)alloc((size_t)NT * HID * 4);
  float*  hv     = (float*)alloc((size_t)NV * HID * 4);
  float*  HSb    = (float*)alloc((size_t)NT * HID * 4);
  float*  HSs    = (float*)alloc((size_t)NV * HID * 4);
  float*  U_task = (float*)alloc((size_t)NT * HID * 4);
  float*  U_vm   = (float*)alloc((size_t)NV * HID * 4);
  __bf16* Hb_t   = (__bf16*)alloc((size_t)NT * HID * 2);
  __bf16* Hb_v   = (__bf16*)alloc((size_t)NV * HID * 2);
  __bf16* Wpk0   = (__bf16*)alloc((size_t)HID * HID * 2);
  __bf16* Wpk1   = (__bf16*)alloc((size_t)HID * HID * 2);
  float*  al_s_b = (float*)alloc((size_t)NT * HEADS * 4);
  float*  al_d_b = (float*)alloc((size_t)NT * HEADS * 4);
  float*  al_s_s = (float*)alloc((size_t)NV * HEADS * 4);
  float*  al_d_s = (float*)alloc((size_t)NV * HEADS * 4);
  float*  mbuf   = (float*)alloc((size_t)NT * HEADS * 4);
  float*  dbuf   = (float*)alloc((size_t)NT * HEADS * 4);
  float*  pooled = (float*)alloc((size_t)HID * 4);

  // input transforms
  k_input_task<<<cdiv(NT * HID, 256), 256, 0, stream>>>(x_task, Wt_in, bt, ht);
  k_input_vm  <<<cdiv(NV * HID, 256), 256, 0, stream>>>(x_vm, Wv_in, bv, hv);

  // one GAT conv: pack weights, WMMA GEMM(s), logits, 3-pass edge softmax-scatter
  auto launch_conv = [&](const __bf16* HbS, int Ns, const __bf16* HbD, int Nd,
                         const float* Wsrc, const float* Wdst, bool shared,
                         const float* avs, const float* avd,
                         const int* esrc, const int* edst, int E, int nLoop,
                         float* HSsrc, float* HSdst, float* als, float* ald,
                         float* U) {
    k_pack_w<<<HID * HID / 256, 256, 0, stream>>>(Wsrc, Wpk0);
    k_gemm_wmma<<<Ns / 16, 128, 0, stream>>>(HbS, Wpk0, HSsrc);
    const float* HSd_use = HSsrc;
    if (!shared) {
      k_pack_w<<<HID * HID / 256, 256, 0, stream>>>(Wdst, Wpk1);
      k_gemm_wmma<<<Nd / 16, 128, 0, stream>>>(HbD, Wpk1, HSdst);
      HSd_use = HSdst;
    }
    k_attn<<<cdiv(Ns * HEADS, 256), 256, 0, stream>>>(HSsrc, avs, als, Ns);
    k_attn<<<cdiv(Nd * HEADS, 256), 256, 0, stream>>>(HSd_use, avd, ald, Nd);
    k_fill<<<cdiv(Nd * HEADS, 256), 256, 0, stream>>>(mbuf, -__builtin_inff(), Nd * HEADS);
    k_fill<<<cdiv(Nd * HEADS, 256), 256, 0, stream>>>(dbuf, 0.f, Nd * HEADS);
    int tot = (E + nLoop) * HEADS;
    k_edge_max    <<<cdiv(tot, 256), 256, 0, stream>>>(esrc, edst, E, nLoop, als, ald, mbuf);
    k_edge_expsum <<<cdiv(tot, 256), 256, 0, stream>>>(esrc, edst, E, nLoop, als, ald, mbuf, dbuf);
    k_edge_scatter<<<cdiv(tot * 4, 256), 256, 0, stream>>>(esrc, edst, E, nLoop, als, ald, mbuf, dbuf, HSsrc, U);
  };

  for (int l = 0; l < LAYERS; ++l) {
    size_t wOff = (size_t)l * HID * HID;
    size_t aOff = (size_t)l * HEADS * CPH;
    size_t bOff = (size_t)l * HID;

    k_f32_to_bf16<<<cdiv(NT * HID, 256), 256, 0, stream>>>(ht, Hb_t, NT * HID);
    k_f32_to_bf16<<<cdiv(NV * HID, 256), 256, 0, stream>>>(hv, Hb_v, NV * HID);
    k_init3<<<cdiv(NT * HID, 256), 256, 0, stream>>>(U_task, dep_b + bOff, rev_b + bOff, rcan_b + bOff, NT);
    k_init1<<<cdiv(NV * HID, 256), 256, 0, stream>>>(U_vm, can_b + bOff, NV);

    // dep: task->task, self loops, shared W
    launch_conv(Hb_t, NT, Hb_t, NT, dep_W + wOff, dep_W + wOff, true,
                dep_as + aOff, dep_ad + aOff, d_src, d_dst, EDEP, NT,
                HSb, HSb, al_s_b, al_d_b, U_task);
    // rev: reversed edges, self loops, shared W
    launch_conv(Hb_t, NT, Hb_t, NT, rev_W + wOff, rev_W + wOff, true,
                rev_as + aOff, rev_ad + aOff, d_dst, d_src, EDEP, NT,
                HSb, HSb, al_s_b, al_d_b, U_task);
    // rcan: vm -> task (src indices are VM ids), no self loops
    launch_conv(Hb_v, NV, Hb_t, NT, rcan_Ws + wOff, rcan_Wd + wOff, false,
                rcan_as + aOff, rcan_ad + aOff, can_dst, can_src, ECAN, 0,
                HSs, HSb, al_s_s, al_d_b, U_task);
    // can: task -> vm, no self loops
    launch_conv(Hb_t, NT, Hb_v, NV, can_Ws + wOff, can_Wd + wOff, false,
                can_as + aOff, can_ad + aOff, can_src, can_dst, ECAN, 0,
                HSb, HSs, al_s_b, al_d_s, U_vm);

    k_combine<<<cdiv(NT * HID, 256), 256, 0, stream>>>(ht, U_task, NT * HID);
    k_combine<<<cdiv(NV * HID, 256), 256, 0, stream>>>(hv, U_vm, NV * HID);
  }

  // mean pool + head MLP
  k_fill<<<1, 256, 0, stream>>>(pooled, 0.f, HID);
  k_pool<<<cdiv(NT, 128), 256, 0, stream>>>(ht, pooled);
  k_mlp<<<1, 128, 0, stream>>>(pooled, W1, b1, W2, b2, (float*)d_out);
}
